// LocalFusion_30176440222533
// MI455X (gfx1250) — matrix operand order repaired and checked by reading, not compiled, not run on hardware
//
#include <hip/hip_runtime.h>
#include <hip/hip_bf16.h>

typedef _Float16 half_t;
typedef __attribute__((ext_vector_type(16))) _Float16 v16h;
typedef __attribute__((ext_vector_type(8)))  _Float16 v8h;
typedef __attribute__((ext_vector_type(8)))  float    v8f;

#define HH 256
#define WW 256
#define NPIX (HH*WW)
#define NIMG 16

// ----------------------------------------------------------------------------
// Pack OIHW f32 conv weights into WMMA B-fragment layout (f16).
// B is K(32) x N(16) per chunk. Lane = n + 16*hi holds column n, K in table
// order (e<8: hi*8+e; e>=8: 16+hi*8+(e-8)) as 16 halfs:
//   dst[chunk*512 + lane*16 + e],  gk = chunk*32 + (e>>3)*16 + hi*8 + (e&7),  gk = cin*9 + tap
// ----------------------------------------------------------------------------
__global__ __launch_bounds__(256) void pack_weights(const float* __restrict__ w,
                                                    half_t* __restrict__ dst,
                                                    int cin, int cout, int nchunks) {
  int i = blockIdx.x * 256 + threadIdx.x;
  int total = nchunks * 512;
  if (i >= total) return;
  int chunk = i >> 9;
  int lane  = (i >> 4) & 31;
  int e     = i & 15;
  int n  = lane & 15;
  int hi = lane >> 4;
  int gk = chunk * 32 + (e >> 3) * 16 + hi * 8 + (e & 7);
  int ktot = cin * 9;
  half_t val = (half_t)0.f;
  if (n < cout && gk < ktot) {
    int ci  = gk / 9;
    int tap = gk - ci * 9;
    val = (half_t)w[(n * cin + ci) * 9 + tap];
  }
  dst[i] = val;
}

// ----------------------------------------------------------------------------
// 3x3 dilated conv as implicit GEMM on WMMA f16->f32.
//   Per wave: 32 pixels (two M=16 tiles sharing one B fragment) x 16 out chans.
//   K = Cin*9 padded to multiples of 32. Block: 8 waves = 8 rows x 32 cols.
// All A-gather LDS offsets are compile-time constants except a per-lane base
// and a 2-way (hi) select of constants -> no runtime div, no EXEC branches.
// Optional epilogue: ReLU then eval-mode BN affine (y = relu(x)*inv + sh).
// ----------------------------------------------------------------------------
template<typename TIN, int CIN, int COUT, int DIL, bool BNRELU>
__global__ __launch_bounds__(256) void conv3x3_wmma(
    const TIN* __restrict__ in,       // (NIMG, CIN, H, W)
    half_t* __restrict__ out,         // plane index = img*out_cstride + n
    const half_t* __restrict__ wpk,   // packed weights for this layer
    const float* __restrict__ bn_g, const float* __restrict__ bn_b,
    const float* __restrict__ bn_m, const float* __restrict__ bn_v,
    int out_cstride) {
  constexpr int TH   = 8 + 2 * DIL;
  constexpr int TW   = 32 + 2 * DIL;
  constexpr int KTOT = CIN * 9;
  constexpr int NCH  = (KTOT + 31) / 32;
  __shared__ half_t smem[CIN * TH * TW];

  const int tid  = threadIdx.x;
  const int lane = tid & 31;
  const int wv   = tid >> 5;            // wave id = row within tile (0..7)
  const int img  = blockIdx.z;
  const int gx0  = blockIdx.x * 32;
  const int gy0  = blockIdx.y * 8;

  // Cooperative load of input tile + halo into LDS (zero-padded borders).
  const size_t in_img = (size_t)img * CIN * NPIX;
  for (int i = tid; i < CIN * TH * TW; i += 256) {
    int c  = i / (TH * TW);
    int r  = i - c * (TH * TW);
    int ly = r / TW;
    int lx = r - ly * TW;
    int gy = gy0 - DIL + ly;
    int gx = gx0 - DIL + lx;
    half_t v = (half_t)0.f;
    if (gy >= 0 && gy < HH && gx >= 0 && gx < WW)
      v = (half_t)in[in_img + (size_t)c * NPIX + gy * WW + gx];
    smem[i] = v;
  }
  __syncthreads();

  const int m    = lane & 15;           // A-row M within tile
  const int hi   = lane >> 4;
  const int base = wv * TW + m;         // per-lane spatial base in LDS

  v8f acc0 = {};
  v8f acc1 = {};
  #pragma unroll
  for (int kc = 0; kc < NCH; ++kc) {
    v16h a0, a1;
    #pragma unroll
    for (int e = 0; e < 16; ++e) {
      // k for hi=0 lanes; hi=1 lanes use k+8. All constants fold after unroll.
      const int gk0 = kc * 32 + ((e >> 3) * 16) + (e & 7);
      const int gk1 = gk0 + 8;
      const int ci0 = gk0 / 9, tp0 = gk0 - ci0 * 9;
      const int ci1 = gk1 / 9, tp1 = gk1 - ci1 * 9;
      const int c0  = (ci0 * TH + (tp0 / 3) * DIL) * TW + (tp0 % 3) * DIL;
      const int c1  = (ci1 * TH + (tp1 / 3) * DIL) * TW + (tp1 % 3) * DIL;
      if (gk1 < KTOT) {                 // both lane-halves valid
        const int off = base + (hi ? c1 : c0);
        a0[e] = smem[off];
        a1[e] = smem[off + 16];
      } else if (gk0 < KTOT) {          // K tail: only hi=0 valid
        half_t t0 = smem[base + c0];
        half_t t1 = smem[base + c0 + 16];
        a0[e] = hi ? (half_t)0.f : t0;
        a1[e] = hi ? (half_t)0.f : t1;
      } else {                          // fully padded
        a0[e] = (half_t)0.f;
        a1[e] = (half_t)0.f;
      }
    }
    // B fragment: one contiguous 32B vector load per lane (pre-packed),
    // shared by both M-tiles.
    v16h b = *(const v16h*)(wpk + kc * 512 + lane * 16);
    acc0 = __builtin_amdgcn_wmma_f32_16x16x32_f16(false, a0, false, b,
                                                  (short)0, acc0, false, false);
    acc1 = __builtin_amdgcn_wmma_f32_16x16x32_f16(false, a1, false, b,
                                                  (short)0, acc1, false, false);
  }

  // D layout: lane holds channel n = lane&15, pixels M = r + 8*hi (r = 0..7).
  const int n = lane & 15;
  if (n < COUT) {
    float inv = 1.f, sh = 0.f;
    if constexpr (BNRELU) {
      float vv = bn_v[n];
      inv = bn_g[n] * rsqrtf(vv + 1e-5f);
      sh  = bn_b[n] - bn_m[n] * inv;
    }
    v8h o0, o1;
    #pragma unroll
    for (int r = 0; r < 8; ++r) {
      float t0 = acc0[r];
      float t1 = acc1[r];
      if constexpr (BNRELU) {
        t0 = fmaxf(t0, 0.f) * inv + sh;
        t1 = fmaxf(t1, 0.f) * inv + sh;
      }
      o0[r] = (half_t)t0;
      o1[r] = (half_t)t1;
    }
    half_t* p = out + ((size_t)img * out_cstride + n) * NPIX
                    + (gy0 + wv) * WW + gx0 + hi * 8;
    *(v8h*)p        = o0;   // 16B stores, naturally aligned
    *(v8h*)(p + 16) = o1;
  }
}

// ----------------------------------------------------------------------------
// Fused: minGRU fwd over V -> minGRU bwd over V (the flip(H) pair cancels,
// since the scan is pointwise in (h,w)) -> alpha 1x1 conv -> softmax over V
// -> weighted sum of raw projected features. One thread per (b,h,w) pixel.
// ----------------------------------------------------------------------------
__device__ __forceinline__ float sigmoidf_(float x) { return 1.f / (1.f + __expf(-x)); }
__device__ __forceinline__ float gact_(float x)     { return x >= 0.f ? x + 0.5f : sigmoidf_(x); }

__global__ __launch_bounds__(256) void gru_alpha_fuse(
    const half_t* __restrict__ fs,    // (16,16,H,W) f16
    const float*  __restrict__ prj,   // (2,8,5,H,W) f32
    const float*  __restrict__ w_f,   // (32,16)
    const float*  __restrict__ w_b,   // (32,16)
    const float*  __restrict__ a_w,   // (16)
    const float*  __restrict__ a_b,   // (1)
    float* __restrict__ outp) {       // (2,5,H,W)
  __shared__ float swf[512], swb[512], saw[16];
  __shared__ float sab;
  for (int i = threadIdx.x; i < 512; i += 256) { swf[i] = w_f[i]; swb[i] = w_b[i]; }
  if (threadIdx.x < 16) saw[threadIdx.x] = a_w[threadIdx.x];
  if (threadIdx.x == 0) sab = a_b[0];
  __syncthreads();

  int gid = blockIdx.x * 256 + threadIdx.x;
  if (gid >= 2 * NPIX) return;
  int b  = gid >> 16;                  // NPIX == 65536
  int hw = gid & (NPIX - 1);

  float h1[8][16];
  float h[16];
  #pragma unroll
  for (int d = 0; d < 16; ++d) h[d] = 0.f;

  // forward minGRU over views
  for (int v = 0; v < 8; ++v) {
    float xv[16];
    const half_t* fp = fs + ((size_t)(b * 8 + v) * 16) * NPIX + hw;
    #pragma unroll
    for (int d = 0; d < 16; ++d) xv[d] = (float)fp[(size_t)d * NPIX];
    #pragma unroll
    for (int e = 0; e < 16; ++e) {
      float z = 0.f, t = 0.f;
      #pragma unroll
      for (int d = 0; d < 16; ++d) {
        z += xv[d] * swf[e * 16 + d];
        t += xv[d] * swf[(16 + e) * 16 + d];
      }
      float zz = sigmoidf_(z);
      float ht = gact_(t);
      h[e] = (1.f - zz) * h[e] + zz * ht;
      h1[v][e] = h[e];
    }
  }

  // backward-branch minGRU (flips over H cancel) + alpha dot, streamed
  float av[8];
  float h2[16];
  #pragma unroll
  for (int d = 0; d < 16; ++d) h2[d] = 0.f;
  for (int v = 0; v < 8; ++v) {
    float acc_a = sab;
    #pragma unroll
    for (int e = 0; e < 16; ++e) {
      float z = 0.f, t = 0.f;
      #pragma unroll
      for (int d = 0; d < 16; ++d) {
        z += h1[v][d] * swb[e * 16 + d];
        t += h1[v][d] * swb[(16 + e) * 16 + d];
      }
      float zz = sigmoidf_(z);
      float ht = gact_(t);
      h2[e] = (1.f - zz) * h2[e] + zz * ht;
      acc_a += h2[e] * saw[e];
    }
    av[v] = acc_a;
  }

  // softmax over V, then weighted sum of projected features
  float mx = av[0];
  #pragma unroll
  for (int v = 1; v < 8; ++v) mx = fmaxf(mx, av[v]);
  float s = 0.f;
  #pragma unroll
  for (int v = 0; v < 8; ++v) { av[v] = __expf(av[v] - mx); s += av[v]; }
  float rs = 1.f / s;
  #pragma unroll
  for (int c = 0; c < 5; ++c) {
    float acc = 0.f;
    #pragma unroll
    for (int v = 0; v < 8; ++v)
      acc += prj[(((size_t)(b * 8 + v)) * 5 + c) * NPIX + hw] * av[v];
    outp[((size_t)b * 5 + c) * NPIX + hw] = acc * rs;
  }
}

// ----------------------------------------------------------------------------
// Launch
// ----------------------------------------------------------------------------
extern "C" void kernel_launch(void* const* d_in, const int* in_sizes, int n_in,
                              void* d_out, int out_size, void* d_ws, size_t ws_size,
                              hipStream_t stream) {
  (void)in_sizes; (void)n_in; (void)out_size; (void)ws_size;

  const float* prj = (const float*)d_in[0];
  // per-layer params: (w,g,b,m,v) x 8 layers starting at index 1
  const float *W[9], *G[8], *Bb[8], *M[8], *V[8];
  int idx = 1;
  for (int l = 0; l < 8; ++l) {
    W[l]  = (const float*)d_in[idx++];
    G[l]  = (const float*)d_in[idx++];
    Bb[l] = (const float*)d_in[idx++];
    M[l]  = (const float*)d_in[idx++];
    V[l]  = (const float*)d_in[idx++];
  }
  W[8] = (const float*)d_in[41];                       // c30_w
  const float* gru_w   = (const float*)d_in[42];
  const float* gru_bw  = (const float*)d_in[43];
  const float* alpha_w = (const float*)d_in[44];
  const float* alpha_b = (const float*)d_in[45];

  // workspace layout (bytes)
  char* ws = (char*)d_ws;
  half_t* wpk    = (half_t*)(ws + 0);                             // ~50 KB packed
  half_t* bufA   = (half_t*)(ws + 65536);                         // 16ch ping 33.5 MB
  half_t* bufB   = (half_t*)(ws + 65536 + 33554432);              // 16ch pong
  half_t* bufCat = (half_t*)(ws + 65536 + 2ull * 33554432);       // 48ch 100.7 MB
  half_t* bufFs  = bufA;                                          // reuse after c23

  dim3 cgrid(WW / 32, HH / 8, NIMG);
  dim3 cblk(256);

  // 1) pack weights (offsets in halfs; chunk counts per layer)
  const int poff[9] = {0, 1024, 2560, 5120, 7680, 10240, 12800, 15360, 17920};
  const int pcin[9] = {5, 8, 16, 16, 16, 16, 16, 16, 48};
  const int pcou[9] = {8, 16, 16, 16, 16, 16, 16, 16, 16};
  const int pnch[9] = {2, 3, 5, 5, 5, 5, 5, 5, 14};
  for (int l = 0; l < 9; ++l) {
    int total = pnch[l] * 512;
    pack_weights<<<(total + 255) / 256, 256, 0, stream>>>(
        W[l], wpk + poff[l], pcin[l], pcou[l], pnch[l]);
  }

  // 2) shallow net (c00 reads the raw f32 input and converts in-staging)
  conv3x3_wmma<float, 5,  8, 1, true ><<<cgrid, cblk, 0, stream>>>(prj,  bufA,
      wpk + poff[0], G[0], Bb[0], M[0], V[0], 8);                  // c00
  conv3x3_wmma<half_t,8, 16, 1, true ><<<cgrid, cblk, 0, stream>>>(bufA, bufB,
      wpk + poff[1], G[1], Bb[1], M[1], V[1], 16);                 // c01
  conv3x3_wmma<half_t,16,16, 2, true ><<<cgrid, cblk, 0, stream>>>(bufB, bufA,
      wpk + poff[2], G[2], Bb[2], M[2], V[2], 16);                 // c02 -> x0_2 (bufA)
  conv3x3_wmma<half_t,16,16, 1, true ><<<cgrid, cblk, 0, stream>>>(bufA, bufCat,
      wpk + poff[3], G[3], Bb[3], M[3], V[3], 48);                 // c03 -> cat[0:16]
  conv3x3_wmma<half_t,16,16, 3, true ><<<cgrid, cblk, 0, stream>>>(bufA, bufB,
      wpk + poff[4], G[4], Bb[4], M[4], V[4], 16);                 // c12
  conv3x3_wmma<half_t,16,16, 1, true ><<<cgrid, cblk, 0, stream>>>(bufB,
      bufCat + (size_t)16 * NPIX,
      wpk + poff[5], G[5], Bb[5], M[5], V[5], 48);                 // c13 -> cat[16:32]
  conv3x3_wmma<half_t,16,16, 4, true ><<<cgrid, cblk, 0, stream>>>(bufA, bufB,
      wpk + poff[6], G[6], Bb[6], M[6], V[6], 16);                 // c22
  conv3x3_wmma<half_t,16,16, 1, true ><<<cgrid, cblk, 0, stream>>>(bufB,
      bufCat + (size_t)32 * NPIX,
      wpk + poff[7], G[7], Bb[7], M[7], V[7], 48);                 // c23 -> cat[32:48]
  conv3x3_wmma<half_t,48,16, 1, false><<<cgrid, cblk, 0, stream>>>(bufCat, bufFs,
      wpk + poff[8], nullptr, nullptr, nullptr, nullptr, 16);      // c30 (no relu/bn)

  // 3) fused GRU fwd/bwd + alpha + softmax + weighted feature sum
  {
    int n = 2 * NPIX;
    gru_alpha_fuse<<<(n + 255) / 256, 256, 0, stream>>>(
        bufFs, prj, gru_w, gru_bw, alpha_w, alpha_b, (float*)d_out);
  }
}
